// DeterministicTrajectoryPredictor_22265110462777
// MI455X (gfx1250) — matrix operand order repaired and checked by reading, not compiled, not run on hardware
//
#include <hip/hip_runtime.h>
#include <hip/hip_bf16.h>
#include <math.h>

// ---------------------------------------------------------------------------
// GRU seq2seq trajectory predictor for MI455X (gfx1250, wave32, WMMA).
// Dominant work: (4096x512)@(512x1536) recurrent GEMMs -> v_wmma_f32_16x16x32_f16.
// Embedding (K=2) folded into fused input-gate weights F = Wih @ W_embed.
// Encoder layers interleaved per time step; each WG owns 16 rows -> in-place
// hidden-state update is race-free (stash in regs/LDS, barrier, write back).
// ---------------------------------------------------------------------------

typedef __attribute__((ext_vector_type(16))) _Float16 v16h;
typedef __attribute__((ext_vector_type(8)))  _Float16 v8h;
typedef __attribute__((ext_vector_type(8)))  float    v8f;

#define HDIM    512
#define G3      1536
#define NROWS   4096
#define ODIM    128
#define TSTEPS  64
#define HORIZON 32
#define LDSLD   (HDIM + 8)   // padded LDS stride (halves), keeps 16B alignment

// ---- WMMA helpers ---------------------------------------------------------

static __device__ __forceinline__ v8f wmma32(v16h a, v16h b, v8f c) {
  return __builtin_amdgcn_wmma_f32_16x16x32_f16(
      /*neg_a=*/false, a, /*neg_b=*/false, b,
      /*c_mod=*/(short)0, c, /*reuse_a=*/false, /*reuse_b=*/false);
}

// A fragment (16x32, f16, row-major source, ld in halves).
// Lane L: M = L&15. Lanes 0-15 hold K=0..7,16..23 ; lanes 16-31 K=8..15,24..31.
static __device__ __forceinline__ v16h load_a_g(const _Float16* __restrict__ base,
                                                int ld, int lane) {
  int r  = lane & 15;
  int k0 = (lane & 16) ? 8 : 0;
  const _Float16* p = base + r * ld + k0;
  v8h lo = *(const v8h*)(p);
  v8h hi = *(const v8h*)(p + 16);
  v16h f;
#pragma unroll
  for (int i = 0; i < 8; ++i) { f[i] = lo[i]; f[i + 8] = hi[i]; }
  return f;
}

static __device__ __forceinline__ v16h load_a_lds(const _Float16* base,
                                                  int ld, int lane) {
  int r  = lane & 15;
  int k0 = (lane & 16) ? 8 : 0;
  const _Float16* p = base + r * ld + k0;
  v8h lo = *(const v8h*)(p);
  v8h hi = *(const v8h*)(p + 16);
  v16h f;
#pragma unroll
  for (int i = 0; i < 8; ++i) { f[i] = lo[i]; f[i + 8] = hi[i]; }
  return f;
}

// B fragment (32x16, f16). Source row-major weight W[n][k] (ld in halves):
// B[k][n] = W[n][k]. Lane L: N = L&15; lanes 0-15 hold K=0..7,16..23 ; lanes 16-31 K=8..15,24..31.
static __device__ __forceinline__ v16h load_b_g(const _Float16* __restrict__ base,
                                                int ld, int lane) {
  int r  = lane & 15;
  int k0 = (lane & 16) ? 8 : 0;
  const _Float16* p = base + r * ld + k0;
  v8h lo = *(const v8h*)(p);
  v8h hi = *(const v8h*)(p + 16);
  v16h f;
#pragma unroll
  for (int i = 0; i < 8; ++i) { f[i] = lo[i]; f[i + 8] = hi[i]; }
  return f;
}

static __device__ __forceinline__ float sigm(float x) {
  return 1.0f / (1.0f + expf(-x));
}

// ---- Small prep kernels ---------------------------------------------------

__global__ __launch_bounds__(256) void k_convert_f16(const float* __restrict__ src,
                                                     _Float16* __restrict__ dst, int n) {
  int i = blockIdx.x * 256 + threadIdx.x;
  if (i < n) dst[i] = (_Float16)src[i];
}

__global__ __launch_bounds__(256) void k_zero_f16(_Float16* __restrict__ dst, int n) {
  int i = blockIdx.x * 256 + threadIdx.x;
  if (i < n) dst[i] = (_Float16)0.0f;
}

// F[g][j] = sum_k Wih[g][k] * W_emb[k][j];  bias[g] = bih[g] + sum_k Wih[g][k]*b_emb[k]
__global__ __launch_bounds__(256) void k_fuse_embed(const float* __restrict__ Wih,
                                                    const float* __restrict__ bih,
                                                    const float* __restrict__ Wemb,
                                                    const float* __restrict__ bemb,
                                                    float* __restrict__ F,
                                                    float* __restrict__ bias) {
  int g = blockIdx.x * 256 + threadIdx.x;
  if (g >= G3) return;
  float f0 = 0.f, f1 = 0.f, bb = 0.f;
  for (int k = 0; k < HDIM; ++k) {
    float w = Wih[g * HDIM + k];
    f0 += w * Wemb[k * 2 + 0];
    f1 += w * Wemb[k * 2 + 1];
    bb += w * bemb[k];
  }
  F[2 * g + 0] = f0;
  F[2 * g + 1] = f1;
  bias[g] = bih[g] + bb;
}

// ctx[b][j] = b_ctx[j] + z[b,:] . W_ctx[j,:]   (32 x 512 outputs, K=1024)
__global__ __launch_bounds__(256) void k_ctx(const float* __restrict__ z,
                                             const float* __restrict__ Wc,
                                             const float* __restrict__ bc,
                                             float* __restrict__ ctx) {
  int i = blockIdx.x * 256 + threadIdx.x;
  if (i >= 32 * HDIM) return;
  int b = i / HDIM, j = i % HDIM;
  float acc = bc[j];
  const float* zr = z + b * 1024;
  const float* wr = Wc + j * 1024;
  for (int k = 0; k < 1024; ++k) acc += zr[k] * wr[k];
  ctx[i] = acc;
}

// hdec = h1 + repeat(ctx)
__global__ __launch_bounds__(256) void k_add_ctx(_Float16* __restrict__ h1,
                                                 const float* __restrict__ ctx) {
  int i = blockIdx.x * 256 + threadIdx.x;
  if (i >= NROWS * HDIM) return;
  int row = i >> 9, j = i & 511;
  int b = row >> 7;
  h1[i] = (_Float16)((float)h1[i] + ctx[b * HDIM + j]);
}

// state init = traj[:, T-1]
__global__ __launch_bounds__(256) void k_state_init(const float* __restrict__ traj,
                                                    float* __restrict__ state) {
  int i = blockIdx.x * 256 + threadIdx.x;
  if (i >= NROWS * 2) return;
  int row = i >> 1, c = i & 1;
  int b = row >> 7, o = row & 127;
  state[i] = traj[(((b * TSTEPS) + (TSTEPS - 1)) * ODIM + o) * 2 + c];
}

// ---- Fused encoder step: layer0 + layer1, one time step ------------------
// grid = 256 WGs (16 rows each), 256 threads (8 waves), wave tile = 16x16.

__global__ __launch_bounds__(256)
void k_enc_step(_Float16* __restrict__ h0, _Float16* __restrict__ h1,
                const float* __restrict__ traj,
                const float* __restrict__ F0, const float* __restrict__ bias0,
                const _Float16* __restrict__ Whh0,
                const _Float16* __restrict__ Wih1, const _Float16* __restrict__ Whh1,
                const float* __restrict__ bhh0,
                const float* __restrict__ bih1, const float* __restrict__ bhh1,
                int t) {
  __shared__ __align__(16) _Float16 lds_h0[16 * LDSLD];
  __shared__ float lds_x[16][2];

  const int tid  = threadIdx.x;
  const int lane = tid & 31;
  const int wid  = tid >> 5;
  const int r0   = blockIdx.x * 16;
  const int ncol = lane & 15;
  const int moff = (lane & 16) ? 8 : 0;

  if (tid < 32) {
    int m = tid >> 1, c = tid & 1;
    int row = r0 + m;
    int b = row >> 7, o = row & 127;
    lds_x[m][c] = traj[(((b * TSTEPS) + t) * ODIM + o) * 2 + c];
  }
  __syncthreads();

  // ---------------- Layer 0: gh = h0 @ Whh0^T, fused GRU -> lds_h0 ---------
#pragma unroll
  for (int i = 0; i < 4; ++i) {
    int c = (wid + 8 * i) * 16;  // 0..496
    v8f ar = {0,0,0,0,0,0,0,0}, az = {0,0,0,0,0,0,0,0}, an = {0,0,0,0,0,0,0,0};
    for (int kb = 0; kb < HDIM; kb += 32) {
      v16h a = load_a_g(h0 + r0 * HDIM + kb, HDIM, lane);
      ar = wmma32(a, load_b_g(Whh0 + (c           ) * HDIM + kb, HDIM, lane), ar);
      az = wmma32(a, load_b_g(Whh0 + (c +   HDIM  ) * HDIM + kb, HDIM, lane), az);
      an = wmma32(a, load_b_g(Whh0 + (c + 2 * HDIM) * HDIM + kb, HDIM, lane), an);
    }
#pragma unroll
    for (int v = 0; v < 8; ++v) {
      int m = moff + v;
      int g = c + ncol;
      float x0 = lds_x[m][0], x1 = lds_x[m][1];
      float gr = x0 * F0[2 * g]              + x1 * F0[2 * g + 1]              + bias0[g];
      float gz = x0 * F0[2 * (g + HDIM)]     + x1 * F0[2 * (g + HDIM) + 1]     + bias0[g + HDIM];
      float gn = x0 * F0[2 * (g + 2 * HDIM)] + x1 * F0[2 * (g + 2 * HDIM) + 1] + bias0[g + 2 * HDIM];
      float hr = ar[v] + bhh0[g];
      float hz = az[v] + bhh0[g + HDIM];
      float hn = an[v] + bhh0[g + 2 * HDIM];
      float r = sigm(gr + hr);
      float z = sigm(gz + hz);
      float nn = tanhf(gn + r * hn);
      float hp = (float)h0[(r0 + m) * HDIM + g];
      lds_h0[m * LDSLD + g] = (_Float16)((1.0f - z) * nn + z * hp);
    }
  }
  __syncthreads();

  // write h0_new to global (next step's layer-0 input)
  for (int idx = tid; idx < 16 * (HDIM / 8); idx += 256) {
    int m = idx >> 6, ch = idx & 63;
    *(v8h*)(h0 + (r0 + m) * HDIM + ch * 8) =
        *(const v8h*)(lds_h0 + m * LDSLD + ch * 8);
  }

  // ---------------- Layer 1: gi = h0_new @ Wih1^T, gh = h1 @ Whh1^T --------
  float hsave[4][8];
#pragma unroll
  for (int i = 0; i < 4; ++i) {
    int c = (wid + 8 * i) * 16;
    v8f air = {0,0,0,0,0,0,0,0}, aiz = {0,0,0,0,0,0,0,0}, ain = {0,0,0,0,0,0,0,0};
    v8f ahr = {0,0,0,0,0,0,0,0}, ahz = {0,0,0,0,0,0,0,0}, ahn = {0,0,0,0,0,0,0,0};
    for (int kb = 0; kb < HDIM; kb += 32) {
      v16h a0 = load_a_lds(lds_h0 + kb, LDSLD, lane);
      v16h a1 = load_a_g(h1 + r0 * HDIM + kb, HDIM, lane);
      air = wmma32(a0, load_b_g(Wih1 + (c           ) * HDIM + kb, HDIM, lane), air);
      aiz = wmma32(a0, load_b_g(Wih1 + (c +   HDIM  ) * HDIM + kb, HDIM, lane), aiz);
      ain = wmma32(a0, load_b_g(Wih1 + (c + 2 * HDIM) * HDIM + kb, HDIM, lane), ain);
      ahr = wmma32(a1, load_b_g(Whh1 + (c           ) * HDIM + kb, HDIM, lane), ahr);
      ahz = wmma32(a1, load_b_g(Whh1 + (c +   HDIM  ) * HDIM + kb, HDIM, lane), ahz);
      ahn = wmma32(a1, load_b_g(Whh1 + (c + 2 * HDIM) * HDIM + kb, HDIM, lane), ahn);
    }
#pragma unroll
    for (int v = 0; v < 8; ++v) {
      int m = moff + v;
      int g = c + ncol;
      float r = sigm(air[v] + bih1[g] + ahr[v] + bhh1[g]);
      float z = sigm(aiz[v] + bih1[g + HDIM] + ahz[v] + bhh1[g + HDIM]);
      float nn = tanhf(ain[v] + bih1[g + 2 * HDIM] + r * (ahn[v] + bhh1[g + 2 * HDIM]));
      float hp = (float)h1[(r0 + m) * HDIM + g];
      hsave[i][v] = (1.0f - z) * nn + z * hp;
    }
  }
  __syncthreads();  // all h1 reads done before in-place overwrite
#pragma unroll
  for (int i = 0; i < 4; ++i) {
    int c = (wid + 8 * i) * 16;
#pragma unroll
    for (int v = 0; v < 8; ++v) {
      int m = moff + v;
      h1[(r0 + m) * HDIM + c + ncol] = (_Float16)hsave[i][v];
    }
  }
}

// ---- Decoder GRU step (single layer; gi from state via fused Fd) ----------

__global__ __launch_bounds__(256)
void k_dec_gru(_Float16* __restrict__ h, const float* __restrict__ state,
               const float* __restrict__ Fd, const float* __restrict__ biasd,
               const _Float16* __restrict__ Whh, const float* __restrict__ bhh) {
  __shared__ float lds_x[16][2];
  const int tid  = threadIdx.x;
  const int lane = tid & 31;
  const int wid  = tid >> 5;
  const int r0   = blockIdx.x * 16;
  const int ncol = lane & 15;
  const int moff = (lane & 16) ? 8 : 0;

  if (tid < 32) {
    int m = tid >> 1, c = tid & 1;
    lds_x[m][c] = state[(r0 + m) * 2 + c];
  }
  __syncthreads();

  float hsave[4][8];
#pragma unroll
  for (int i = 0; i < 4; ++i) {
    int c = (wid + 8 * i) * 16;
    v8f ar = {0,0,0,0,0,0,0,0}, az = {0,0,0,0,0,0,0,0}, an = {0,0,0,0,0,0,0,0};
    for (int kb = 0; kb < HDIM; kb += 32) {
      v16h a = load_a_g(h + r0 * HDIM + kb, HDIM, lane);
      ar = wmma32(a, load_b_g(Whh + (c           ) * HDIM + kb, HDIM, lane), ar);
      az = wmma32(a, load_b_g(Whh + (c +   HDIM  ) * HDIM + kb, HDIM, lane), az);
      an = wmma32(a, load_b_g(Whh + (c + 2 * HDIM) * HDIM + kb, HDIM, lane), an);
    }
#pragma unroll
    for (int v = 0; v < 8; ++v) {
      int m = moff + v;
      int g = c + ncol;
      float x0 = lds_x[m][0], x1 = lds_x[m][1];
      float gr = x0 * Fd[2 * g]              + x1 * Fd[2 * g + 1]              + biasd[g];
      float gz = x0 * Fd[2 * (g + HDIM)]     + x1 * Fd[2 * (g + HDIM) + 1]     + biasd[g + HDIM];
      float gn = x0 * Fd[2 * (g + 2 * HDIM)] + x1 * Fd[2 * (g + 2 * HDIM) + 1] + biasd[g + 2 * HDIM];
      float r = sigm(gr + ar[v] + bhh[g]);
      float z = sigm(gz + az[v] + bhh[g + HDIM]);
      float nn = tanhf(gn + r * (an[v] + bhh[g + 2 * HDIM]));
      float hp = (float)h[(r0 + m) * HDIM + g];
      hsave[i][v] = (1.0f - z) * nn + z * hp;
    }
  }
  __syncthreads();
#pragma unroll
  for (int i = 0; i < 4; ++i) {
    int c = (wid + 8 * i) * 16;
#pragma unroll
    for (int v = 0; v < 8; ++v) {
      int m = moff + v;
      h[(r0 + m) * HDIM + c + ncol] = (_Float16)hsave[i][v];
    }
  }
}

// ---- Decoder MLP hidden: mid = gelu(h @ W1^T + b1) -------------------------

__global__ __launch_bounds__(256)
void k_dec_mid(const _Float16* __restrict__ h, const _Float16* __restrict__ W1h,
               const float* __restrict__ b1, _Float16* __restrict__ mid) {
  const int tid  = threadIdx.x;
  const int lane = tid & 31;
  const int wid  = tid >> 5;
  const int r0   = blockIdx.x * 16;
  const int ncol = lane & 15;
  const int moff = (lane & 16) ? 8 : 0;

#pragma unroll
  for (int i = 0; i < 4; ++i) {
    int c = (wid + 8 * i) * 16;
    v8f acc = {0,0,0,0,0,0,0,0};
    for (int kb = 0; kb < HDIM; kb += 32) {
      v16h a = load_a_g(h + r0 * HDIM + kb, HDIM, lane);
      acc = wmma32(a, load_b_g(W1h + c * HDIM + kb, HDIM, lane), acc);
    }
#pragma unroll
    for (int v = 0; v < 8; ++v) {
      int m = moff + v;
      int j = c + ncol;
      float x = acc[v] + b1[j];
      float gl = 0.5f * x * (1.0f + erff(x * 0.70710678118654752f));
      mid[(r0 + m) * HDIM + j] = (_Float16)gl;
    }
  }
}

// ---- Decoder head: delta = mid @ W2^T + b2; tanh-clamp; geo update ---------

__global__ __launch_bounds__(256)
void k_dec_out(const _Float16* __restrict__ mid, const float* __restrict__ W2,
               const float* __restrict__ b2, float* __restrict__ state,
               float* __restrict__ out, int step) {
  int row = blockIdx.x * 256 + threadIdx.x;
  if (row >= NROWS) return;
  float d0 = b2[0], d1 = b2[1];
  const _Float16* mr = mid + row * HDIM;
  for (int k = 0; k < HDIM; ++k) {
    float mv = (float)mr[k];
    d0 += mv * W2[k];
    d1 += mv * W2[HDIM + k];
  }
  d0 = 5.0f * tanhf(d0 * 0.2f);
  d1 = 5.0f * tanhf(d1 * 0.2f);
  float lat = state[2 * row] + d0;
  float lon = state[2 * row + 1] + d1;
  lat = fminf(90.0f, fmaxf(-90.0f, lat));
  lon = fmodf(lon, 360.0f);
  if (lon < 0.0f) lon += 360.0f;
  state[2 * row]     = lat;
  state[2 * row + 1] = lon;
  int b = row >> 7, o = row & 127;
  out[((b * HORIZON + step) * ODIM + o) * 2 + 0] = lat;
  out[((b * HORIZON + step) * ODIM + o) * 2 + 1] = lon;
}

// ---------------------------------------------------------------------------

extern "C" void kernel_launch(void* const* d_in, const int* in_sizes, int n_in,
                              void* d_out, int out_size, void* d_ws, size_t ws_size,
                              hipStream_t stream) {
  (void)in_sizes; (void)n_in; (void)out_size; (void)ws_size;
  const float* z_bg  = (const float*)d_in[0];
  const float* traj  = (const float*)d_in[1];
  // d_in[2] = horizon (device int, fixed at 32 in reference; hardcoded)
  const float* W_emb = (const float*)d_in[3];
  const float* b_emb = (const float*)d_in[4];
  const float* W_ctx = (const float*)d_in[5];
  const float* b_ctx = (const float*)d_in[6];
  const float* eWih  = (const float*)d_in[7];
  const float* eWhh  = (const float*)d_in[8];
  const float* ebih  = (const float*)d_in[9];
  const float* ebhh  = (const float*)d_in[10];
  const float* dWih  = (const float*)d_in[11];
  const float* dWhh  = (const float*)d_in[12];
  const float* dbih  = (const float*)d_in[13];
  const float* dbhh  = (const float*)d_in[14];
  const float* W1    = (const float*)d_in[15];
  const float* b1    = (const float*)d_in[16];
  const float* W2    = (const float*)d_in[17];
  const float* b2    = (const float*)d_in[18];
  float* out = (float*)d_out;

  // workspace layout (bytes)
  char* w = (char*)d_ws;
  _Float16* Whh0h = (_Float16*)(w + 0);         // 1536*512 f16
  _Float16* Wih1h = (_Float16*)(w + 1572864);
  _Float16* Whh1h = (_Float16*)(w + 3145728);
  _Float16* Wdech = (_Float16*)(w + 4718592);
  _Float16* W1h   = (_Float16*)(w + 6291456);   // 512*512 f16
  _Float16* h0    = (_Float16*)(w + 6815744);   // 4096*512 f16
  _Float16* h1    = (_Float16*)(w + 11010048);
  _Float16* mid   = (_Float16*)(w + 15204352);
  float*    F0    = (float*)(w + 19398656);     // 1536*2
  float*    bias0 = (float*)(w + 19410944);     // 1536
  float*    Fd    = (float*)(w + 19417088);
  float*    biasd = (float*)(w + 19429376);
  float*    ctx   = (float*)(w + 19435520);     // 32*512
  float*    state = (float*)(w + 19501056);     // 4096*2

  const int WG3 = G3 * HDIM;  // 786432

  // weight conversion to f16 (weights are hot in L2 thereafter)
  k_convert_f16<<<(WG3 + 255) / 256, 256, 0, stream>>>(eWhh,        Whh0h, WG3);
  k_convert_f16<<<(WG3 + 255) / 256, 256, 0, stream>>>(eWih + WG3,  Wih1h, WG3);
  k_convert_f16<<<(WG3 + 255) / 256, 256, 0, stream>>>(eWhh + WG3,  Whh1h, WG3);
  k_convert_f16<<<(WG3 + 255) / 256, 256, 0, stream>>>(dWhh,        Wdech, WG3);
  k_convert_f16<<<(HDIM * HDIM + 255) / 256, 256, 0, stream>>>(W1,  W1h, HDIM * HDIM);

  k_zero_f16<<<(NROWS * HDIM + 255) / 256, 256, 0, stream>>>(h0, NROWS * HDIM);
  k_zero_f16<<<(NROWS * HDIM + 255) / 256, 256, 0, stream>>>(h1, NROWS * HDIM);

  k_fuse_embed<<<6, 256, 0, stream>>>(eWih, ebih, W_emb, b_emb, F0, bias0);
  k_fuse_embed<<<6, 256, 0, stream>>>(dWih, dbih, W_emb, b_emb, Fd, biasd);
  k_ctx<<<64, 256, 0, stream>>>(z_bg, W_ctx, b_ctx, ctx);

  // encoder: 64 fused (layer0+layer1) WMMA time steps
  for (int t = 0; t < TSTEPS; ++t) {
    k_enc_step<<<NROWS / 16, 256, 0, stream>>>(
        h0, h1, traj, F0, bias0, Whh0h, Wih1h, Whh1h,
        ebhh, ebih + G3, ebhh + G3, t);
  }

  // decoder init: h = h_final + ctx ; state = traj[:, -1]
  k_add_ctx<<<(NROWS * HDIM + 255) / 256, 256, 0, stream>>>(h1, ctx);
  k_state_init<<<(NROWS * 2 + 255) / 256, 256, 0, stream>>>(traj, state);

  // decoder rollout
  for (int s = 0; s < HORIZON; ++s) {
    k_dec_gru<<<NROWS / 16, 256, 0, stream>>>(h1, state, Fd, biasd, Wdech, dbhh);
    k_dec_mid<<<NROWS / 16, 256, 0, stream>>>(h1, W1h, b1, mid);
    k_dec_out<<<NROWS / 256, 256, 0, stream>>>(mid, W2, b2, state, out, s);
  }
}
